// Main_net_18004502904887
// MI455X (gfx1250) — matrix operand order repaired and checked by reading, not compiled, not run on hardware
//
#include <hip/hip_runtime.h>

// MI455X / gfx1250, wave32.  3x3 convs (~165 GFLOP) as implicit-GEMM via
// V_WMMA_F32_16X16X32_F16.  Activations NHWC, channel-padded to CP and
// spatially halo-padded (zero border) so the WMMA K-loop is branch-free:
// A fragment = 2 unconditional b128 loads, B fragment = 1 contiguous 32B read.

typedef __attribute__((ext_vector_type(16))) _Float16 v16h;
typedef __attribute__((ext_vector_type(8)))  _Float16 v8h;
typedef __attribute__((ext_vector_type(8)))  float    v8f;

// ---------------------------------------------------------------------------
// Elementwise / layout kernels
// ---------------------------------------------------------------------------
__global__ void zero_f16_kernel(_Float16* __restrict__ p, long long n) {
  long long i = (long long)blockIdx.x * blockDim.x + threadIdx.x;
  if (i < n) p[i] = (_Float16)0.0f;
}

__global__ void zero_f32_kernel(float* __restrict__ p, long long n) {
  long long i = (long long)blockIdx.x * blockDim.x + threadIdx.x;
  if (i < n) p[i] = 0.0f;
}

// Zero the 1-pixel spatial border of a halo NHWC buffer [B][H+2][W+2][C].
__global__ void zero_halo_kernel(_Float16* __restrict__ buf,
                                 int B, int H, int W, int C) {
  int Hp = H + 2, Wp = W + 2;
  long long nb = 2LL * Wp + 2LL * H;             // border pixels per image
  long long tot = (long long)B * nb * C;
  long long idx = (long long)blockIdx.x * blockDim.x + threadIdx.x;
  if (idx >= tot) return;
  int c = (int)(idx % C);
  long long t = idx / C;
  long long j = t % nb;
  int b = (int)(t / nb);
  int yp, xp;
  if (j < Wp)              { yp = 0;      xp = (int)j; }
  else if (j < 2LL * Wp)   { yp = Hp - 1; xp = (int)(j - Wp); }
  else { long long k = j - 2LL * Wp; yp = 1 + (int)(k >> 1); xp = (k & 1) ? (Wp - 1) : 0; }
  buf[(((long long)b * Hp + yp) * Wp + xp) * C + c] = (_Float16)0.0f;
}

// NCHW f32 3-ch image -> halo NHWC f16, 16-ch zero padded, zero border.
__global__ void img_to_nhwc16_kernel(const float* __restrict__ in,
                                     _Float16* __restrict__ out,
                                     int B, int H, int W) {
  int Hp = H + 2, Wp = W + 2;
  long long np = (long long)H * W;
  long long tot = (long long)B * Hp * Wp;
  long long i = (long long)blockIdx.x * blockDim.x + threadIdx.x;
  if (i >= tot) return;
  int xp = (int)(i % Wp);
  long long t = i / Wp;
  int yp = (int)(t % Hp);
  int b  = (int)(t / Hp);
  _Float16* o = out + i * 16;
  float v[3] = {0.0f, 0.0f, 0.0f};
  if (yp >= 1 && yp <= H && xp >= 1 && xp <= W) {
    long long p = (long long)(yp - 1) * W + (xp - 1);
    #pragma unroll
    for (int c = 0; c < 3; ++c) v[c] = in[((long long)b * 3 + c) * np + p];
  }
  #pragma unroll
  for (int c = 0; c < 3; ++c) o[c] = (_Float16)v[c];
  #pragma unroll
  for (int c = 3; c < 16; ++c) o[c] = (_Float16)0.0f;
}

// Linear NHWC f32 [B,H,W,C] -> halo NHWC f16 (zero border).
__global__ void f32_to_f16_halo_kernel(const float* __restrict__ in,
                                       _Float16* __restrict__ out,
                                       int B, int H, int W, int C) {
  int Hp = H + 2, Wp = W + 2;
  long long tot = (long long)B * Hp * Wp * C;
  long long idx = (long long)blockIdx.x * blockDim.x + threadIdx.x;
  if (idx >= tot) return;
  int c = (int)(idx % C);
  long long t = idx / C;
  int xp = (int)(t % Wp); t /= Wp;
  int yp = (int)(t % Hp);
  int b  = (int)(t / Hp);
  float v = 0.0f;
  if (yp >= 1 && yp <= H && xp >= 1 && xp <= W)
    v = in[(((long long)b * H + (yp - 1)) * W + (xp - 1)) * C + c];
  out[idx] = (_Float16)v;
}

// NCHW f32 [B,2,h,w] flow -> linear NHWC f32 [B,h,w,2]
__global__ void flow_to_nhwc_kernel(const float* __restrict__ in,
                                    float* __restrict__ out,
                                    int B, int H, int W) {
  long long np = (long long)H * W;
  long long i = (long long)blockIdx.x * blockDim.x + threadIdx.x;
  if (i >= (long long)B * np) return;
  long long p = i % np;
  int b = (int)(i / np);
  out[i * 2 + 0] = in[((long long)b * 2 + 0) * np + p];
  out[i * 2 + 1] = in[((long long)b * 2 + 1) * np + p];
}

// Half-pixel bilinear resize, linear NHWC f32, fused scale, optional add.
__global__ void resize_nhwc_kernel(const float* __restrict__ in,
                                   float* __restrict__ out,
                                   int B, int C, int Hin, int Win,
                                   int Hout, int Wout,
                                   float scale, int accumulate) {
  long long idx = (long long)blockIdx.x * blockDim.x + threadIdx.x;
  long long tot = (long long)B * Hout * Wout * C;
  if (idx >= tot) return;
  int c = (int)(idx % C);
  long long t = idx / C;
  int ox = (int)(t % Wout); t /= Wout;
  int oy = (int)(t % Hout);
  int b  = (int)(t / Hout);
  float ry = (float)Hin / (float)Hout;
  float rx = (float)Win / (float)Wout;
  float fy = (oy + 0.5f) * ry - 0.5f;
  float fx = (ox + 0.5f) * rx - 0.5f;
  int y0 = (int)floorf(fy), x0 = (int)floorf(fx);
  float wy = fy - (float)y0, wx = fx - (float)x0;
  int y0c = min(max(y0, 0), Hin - 1), y1c = min(max(y0 + 1, 0), Hin - 1);
  int x0c = min(max(x0, 0), Win - 1), x1c = min(max(x0 + 1, 0), Win - 1);
  const float* p = in + (long long)b * Hin * Win * C + c;
  long long rs = (long long)Win * C;
  float v = (1.0f - wy) * ((1.0f - wx) * p[y0c * rs + (long long)x0c * C] +
                           wx          * p[y0c * rs + (long long)x1c * C]) +
            wy          * ((1.0f - wx) * p[y1c * rs + (long long)x0c * C] +
                           wx          * p[y1c * rs + (long long)x1c * C]);
  v *= scale;
  if (accumulate) out[idx] += v; else out[idx] = v;
}

// Fused backwarp + metric-UNet input builder, writing halo NHWC f16 (16 ch):
// ch0 = mean|imgS - warp(imgO, flow)|, ch1..3 = imgS, ch4..15 = 0, border = 0.
__global__ void build_mu_input_kernel(const float* __restrict__ imgS,
                                      const float* __restrict__ imgO,
                                      const float* __restrict__ flow,
                                      _Float16* __restrict__ out16,
                                      int B, int H, int W) {
  int Hp = H + 2, Wp = W + 2;
  long long np = (long long)H * W;
  long long tot = (long long)B * Hp * Wp;
  long long i = (long long)blockIdx.x * blockDim.x + threadIdx.x;
  if (i >= tot) return;
  int xp = (int)(i % Wp);
  long long t = i / Wp;
  int yp = (int)(t % Hp);
  int b  = (int)(t / Hp);
  _Float16* o = out16 + i * 16;
  #pragma unroll
  for (int c = 0; c < 16; ++c) o[c] = (_Float16)0.0f;
  if (yp < 1 || yp > H || xp < 1 || xp > W) return;
  int x = xp - 1, y = yp - 1;
  long long iflow = ((long long)b * H + y) * W + x;
  float fx = (float)x + flow[iflow * 2 + 0];
  float fy = (float)y + flow[iflow * 2 + 1];
  float x0f = floorf(fx), y0f = floorf(fy);
  int x0 = (int)x0f, y0 = (int)y0f;
  float ax = fx - x0f, ay = fy - y0f;
  int   xs[4] = {x0, x0 + 1, x0, x0 + 1};
  int   ys[4] = {y0, y0, y0 + 1, y0 + 1};
  float ws[4] = {(1.0f - ax) * (1.0f - ay), ax * (1.0f - ay),
                 (1.0f - ax) * ay,          ax * ay};
  float warped[3] = {0.0f, 0.0f, 0.0f};
  #pragma unroll
  for (int k = 0; k < 4; ++k) {
    if (xs[k] >= 0 && xs[k] < W && ys[k] >= 0 && ys[k] < H) {
      long long off = (long long)ys[k] * W + xs[k];
      #pragma unroll
      for (int c = 0; c < 3; ++c)
        warped[c] += imgO[((long long)b * 3 + c) * np + off] * ws[k];
    }
  }
  float d = 0.0f, sv[3];
  #pragma unroll
  for (int c = 0; c < 3; ++c) {
    sv[c] = imgS[((long long)b * 3 + c) * np + (long long)y * W + x];
    d += fabsf(sv[c] - warped[c]);
  }
  o[0] = (_Float16)(d * (1.0f / 3.0f));
  #pragma unroll
  for (int c = 0; c < 3; ++c) o[1 + c] = (_Float16)sv[c];
}

// Softmax splatting scatter: inp = halo NHWC f16 (stride Cs), flow/metric
// linear, acc [B*np][(C+1)] f32, contiguous atomics per corner.
__global__ void softsplat_scatter_kernel(const _Float16* __restrict__ inp,
                                         const float* __restrict__ flow,
                                         const float* __restrict__ metric,
                                         const float* __restrict__ beta,
                                         float* __restrict__ acc,
                                         int B, int C, int Cs, int H, int W) {
  long long np = (long long)H * W;
  long long i = (long long)blockIdx.x * blockDim.x + threadIdx.x;
  if (i >= (long long)B * np) return;
  int x = (int)(i % W);
  long long t = i / W;
  int y = (int)(t % H);
  int b = (int)(t / H);
  float Z = expf(beta[0] * metric[i]);
  float fx = (float)x + flow[i * 2 + 0];
  float fy = (float)y + flow[i * 2 + 1];
  float x0f = floorf(fx), y0f = floorf(fy);
  int x0 = (int)x0f, y0 = (int)y0f;
  float ax = fx - x0f, ay = fy - y0f;
  int   xs[4] = {x0, x0 + 1, x0, x0 + 1};
  int   ys[4] = {y0, y0, y0 + 1, y0 + 1};
  float ww[4] = {(1.0f - ax) * (1.0f - ay), ax * (1.0f - ay),
                 (1.0f - ax) * ay,          ax * ay};
  const _Float16* ip = inp +
      (((long long)b * (H + 2) + (y + 1)) * (W + 2) + (x + 1)) * (long long)Cs;
  #pragma unroll
  for (int k = 0; k < 4; ++k) {
    if (xs[k] >= 0 && xs[k] < W && ys[k] >= 0 && ys[k] < H) {
      long long base = ((long long)b * np + (long long)ys[k] * W + xs[k]) * (C + 1);
      float w = ww[k];
      for (int c = 0; c < C; ++c)
        atomicAdd(acc + base + c, (float)ip[c] * Z * w);
      atomicAdd(acc + base + C, Z * w);
    }
  }
}

// num/den (den==0 -> 1); write f16 channel slice into halo NHWC concat buffer.
__global__ void softsplat_norm_kernel(const float* __restrict__ acc,
                                      _Float16* __restrict__ dst,
                                      int B, int C, int H, int W,
                                      int Ctot, int chOff) {
  long long np = (long long)H * W;
  long long tot = (long long)B * np * C;
  long long idx = (long long)blockIdx.x * blockDim.x + threadIdx.x;
  if (idx >= tot) return;
  int c = (int)(idx % C);
  long long pix = idx / C;                  // b*np + p
  int x = (int)(pix % W);
  long long t = pix / W;
  int y = (int)(t % H);
  int b = (int)(t / H);
  float den = acc[pix * (C + 1) + C];
  den = (den == 0.0f) ? 1.0f : den;
  float num = acc[pix * (C + 1) + c];
  dst[((((long long)b * (H + 2) + (y + 1)) * (W + 2) + (x + 1)) * (long long)Ctot)
      + chOff + c] = (_Float16)(num / den);
}

// ---------------------------------------------------------------------------
// Weight packing: OIHW f32 -> WMMA B-fragment f16, K-order = tap*CP + ci.
// Element (step s, tile t, lane L, half h): K = s*32 + h + 8*(h>=8) + 8*(L>=16),
// N = t*16 + (L&15); zero outside (Cout, 9 taps, Cin).
// ---------------------------------------------------------------------------
__global__ void pack_wfrag_kernel(const float* __restrict__ w,
                                  _Float16* __restrict__ out,
                                  int Cout, int Cin, int CP, int NT,
                                  long long tot) {
  long long idx = (long long)blockIdx.x * blockDim.x + threadIdx.x;
  if (idx >= tot) return;
  int h = (int)(idx & 15);
  long long r = idx >> 4;
  int L = (int)(r & 31);
  r >>= 5;
  int tt = (int)(r % NT);
  int s  = (int)(r / NT);
  int K = s * 32 + h + 8 * (h >> 3) + 8 * (L >> 4);
  int n = tt * 16 + (L & 15);
  int tap = K / CP, ci = K % CP;
  float val = 0.0f;
  if (n < Cout && tap < 9 && ci < Cin)
    val = w[(((long long)n * Cin + ci) * 3 + tap / 3) * 3 + tap % 3];
  out[idx] = (_Float16)val;
}

// ---------------------------------------------------------------------------
// Implicit-GEMM 3x3 conv on halo NHWC f16 input (channel stride CP).
// One wave -> 16 output pixels x (NT*16) channels.  Branch-free MMA loop:
// halo makes every A load in-bounds; zero-padded weights absorb K padding.
// outMode: 0 = linear NHWC f32/f16, 1 = halo NHWC f16, 2 = NCHW f32.
// ---------------------------------------------------------------------------
template <int CP, int NT>
__global__ __launch_bounds__(128)
void conv3x3_wmma_kernel(const _Float16* __restrict__ in,
                         const _Float16* __restrict__ wfrag,
                         const float* __restrict__ bias,
                         float* __restrict__ outF,
                         _Float16* __restrict__ outH,
                         int B, int Hin, int Win,
                         int Cout, int CoutStride, int Hout, int Wout,
                         int stride, int padBeg, int relu, int outMode) {
  constexpr int KSTEPS = (9 * CP + 31) / 32;
  int lane = threadIdx.x & 31;
  int wave = threadIdx.x >> 5;
  long long Npix = (long long)B * Hout * Wout;
  long long m0 = ((long long)blockIdx.x * 4 + wave) * 16;
  if (m0 >= Npix) return;                    // wave-uniform

  long long mrow = m0 + (lane & 15);
  bool mvalid = mrow < Npix;
  int b = 0, oy = 0, ox = 0;
  {
    long long t = mvalid ? mrow : 0;
    ox = (int)(t % Wout); t /= Wout;
    oy = (int)(t % Hout);
    b  = (int)(t / Hout);
  }
  int group = lane >> 4;
  int Hp = Hin + 2, Wp = Win + 2;
  int iy0 = oy * stride - padBeg + 1;        // halo-adjusted top-left tap row
  int ix0 = ox * stride - padBeg + 1;
  const _Float16* inb = in + (long long)b * Hp * Wp * CP + 8 * group;

  v8f zero8 = {};
  v8f acc[NT];
  #pragma unroll
  for (int t = 0; t < NT; ++t) acc[t] = zero8;

  if constexpr (CP >= 32) {
    constexpr int CB = CP / 32;
    int s = 0;
    #pragma unroll
    for (int ky = 0; ky < 3; ++ky) {
      #pragma unroll
      for (int kx = 0; kx < 3; ++kx) {
        const _Float16* base =
            inb + ((long long)(iy0 + ky) * Wp + (ix0 + kx)) * CP;
        #pragma unroll
        for (int cb = 0; cb < CB; ++cb, ++s) {
          v8h c0 = *(const v8h*)(base + cb * 32);
          v8h c1 = *(const v8h*)(base + cb * 32 + 16);
          v16h a = __builtin_shufflevector(c0, c1, 0, 1, 2, 3, 4, 5, 6, 7,
                                           8, 9, 10, 11, 12, 13, 14, 15);
          const _Float16* wp = wfrag + (((long long)s * NT) * 32 + lane) * 16;
          #pragma unroll
          for (int t = 0; t < NT; ++t) {
            v16h bf = *(const v16h*)(wp + (long long)t * 512);
            acc[t] = __builtin_amdgcn_wmma_f32_16x16x32_f16(
                false, a, false, bf, (short)0, acc[t], false, false);
          }
        }
      }
    }
  } else {
    #pragma unroll
    for (int s = 0; s < KSTEPS; ++s) {
      v16h a;
      #pragma unroll
      for (int j = 0; j < 2; ++j) {
        int Kc = s * 32 + 16 * j;            // +8*group folded into inb
        int tap = Kc / CP; if (tap > 8) tap = 8;   // padded K -> B is zero
        int ci  = Kc % CP;
        int ky = tap / 3, kx = tap % 3;
        v8h chunk = *(const v8h*)(
            inb + ((long long)(iy0 + ky) * Wp + (ix0 + kx)) * CP + ci);
        #pragma unroll
        for (int e = 0; e < 8; ++e) a[j * 8 + e] = chunk[e];
      }
      const _Float16* wp = wfrag + (((long long)s * NT) * 32 + lane) * 16;
      #pragma unroll
      for (int t = 0; t < NT; ++t) {
        v16h bf = *(const v16h*)(wp + (long long)t * 512);
        acc[t] = __builtin_amdgcn_wmma_f32_16x16x32_f16(
            false, a, false, bf, (short)0, acc[t], false, false);
      }
    }
  }

  // D store: lane holds col n = t*16 + (lane&15), rows m0 + r + 8*group.
  #pragma unroll
  for (int t = 0; t < NT; ++t) {
    int n = t * 16 + (lane & 15);
    if (n < Cout) {
      float bv = bias ? bias[n] : 0.0f;
      #pragma unroll
      for (int r = 0; r < 8; ++r) {
        long long m = m0 + r + 8 * group;
        if (m < Npix) {
          long long tt = m;
          int sx = (int)(tt % Wout); tt /= Wout;
          int sy = (int)(tt % Hout);
          int sb = (int)(tt / Hout);
          float vv = acc[t][r] + bv;
          if (relu) vv = fmaxf(vv, 0.0f);
          long long oidx;
          if (outMode == 2)
            oidx = (((long long)sb * Cout + n) * Hout + sy) * Wout + sx;
          else if (outMode == 1)
            oidx = (((long long)sb * (Hout + 2) + sy + 1) * (Wout + 2) + sx + 1)
                   * (long long)CoutStride + n;
          else
            oidx = (((long long)sb * Hout + sy) * Wout + sx)
                   * (long long)CoutStride + n;
          if (outH) outH[oidx] = (_Float16)vv;
          else      outF[oidx] = vv;
        }
      }
    }
  }
}

// ---------------------------------------------------------------------------
// Host orchestration
// ---------------------------------------------------------------------------
struct ConvW { const _Float16* wf; int CP; int NT; int Cout; };

extern "C" void kernel_launch(void* const* d_in, const int* in_sizes, int n_in,
                              void* d_out, int out_size, void* d_ws, size_t ws_size,
                              hipStream_t stream) {
  (void)in_sizes; (void)n_in; (void)out_size; (void)ws_size;

  const int B = 2, H = 512, W = 896;
  const int H2 = 256, W2 = 448, H4 = 128, W4 = 224;
  const long long N   = (long long)H  * W;
  const long long Nh  = (long long)H2 * W2;
  const long long Nq  = (long long)H4 * W4;
  const long long Np  = (long long)(H  + 2) * (W  + 2);   // halo pixel counts
  const long long Nhp = (long long)(H2 + 2) * (W2 + 2);
  const long long Nqp = (long long)(H4 + 2) * (W4 + 2);

  const float* img1   = (const float*)d_in[0];
  const float* img2   = (const float*)d_in[1];
  const float* flow12 = (const float*)d_in[2];
  const float* flow21 = (const float*)d_in[3];
  const float* fe1w[3] = {(const float*)d_in[4],  (const float*)d_in[6],  (const float*)d_in[8]};
  const float* fe1b[3] = {(const float*)d_in[5],  (const float*)d_in[7],  (const float*)d_in[9]};
  const float* fe2w[3] = {(const float*)d_in[10], (const float*)d_in[12], (const float*)d_in[14]};
  const float* fe2b[3] = {(const float*)d_in[11], (const float*)d_in[13], (const float*)d_in[15]};
  const float* mu_w1 = (const float*)d_in[16];
  const float* mu_b1 = (const float*)d_in[17];
  const float* mu_w2 = (const float*)d_in[18];
  const float* mu_b2 = (const float*)d_in[19];
  const float* gn_w3 = (const float*)d_in[20];
  const float* gn_b3 = (const float*)d_in[21];
  const float* gn_w2 = (const float*)d_in[22];
  const float* gn_b2 = (const float*)d_in[23];
  const float* gn_w1 = (const float*)d_in[24];
  const float* gn_b1 = (const float*)d_in[25];
  const float* gn_wf = (const float*)d_in[26];
  const float* gn_bf = (const float*)d_in[27];
  const float* beta1 = (const float*)d_in[28];
  const float* beta2 = (const float*)d_in[29];

  size_t cur = 0;
  auto alloc = [&](size_t bytes) -> void* {
    cur = (cur + 255) & ~(size_t)255;
    void* p = (char*)d_ws + cur;
    cur += bytes;
    return p;
  };
  auto blocks = [](long long n) { return dim3((unsigned)((n + 255) / 256)); };

  auto packw = [&](const float* w, int Cout, int Cin, int CP) -> ConvW {
    int NT = (Cout + 15) / 16;
    int ksteps = (9 * CP + 31) / 32;
    long long tot = (long long)ksteps * NT * 512;
    _Float16* buf = (_Float16*)alloc((size_t)tot * 2);
    pack_wfrag_kernel<<<blocks(tot), 256, 0, stream>>>(w, buf, Cout, Cin, CP, NT, tot);
    return {buf, CP, NT, Cout};
  };
  ConvW cw_fe1[3] = {packw(fe1w[0], 32, 3, 16), packw(fe1w[1], 64, 32, 32), packw(fe1w[2], 96, 64, 64)};
  ConvW cw_fe2[3] = {packw(fe2w[0], 32, 3, 16), packw(fe2w[1], 64, 32, 32), packw(fe2w[2], 96, 64, 64)};
  ConvW cw_mu1 = packw(mu_w1, 16, 4, 16);
  ConvW cw_mu2 = packw(mu_w2, 1, 16, 16);
  ConvW cw_gn3 = packw(gn_w3, 64, 192, 192);
  ConvW cw_gn2 = packw(gn_w2, 64, 128, 128);
  ConvW cw_gn1 = packw(gn_w1, 64, 70, 96);
  ConvW cw_gnf = packw(gn_wf, 3, 64, 64);

  auto conv = [&](const _Float16* in, const ConvW& cw, const float* bias,
                  float* outF, _Float16* outH, int coutStride,
                  int Hin, int Win, int Hout, int Wout, int stride, int relu, int outMode) {
    int padBeg = (stride == 1) ? 1 : 0;
    long long Npix = (long long)B * Hout * Wout;
    dim3 g((unsigned)((Npix + 63) / 64));
    #define CONV_CASE(cp, nt)                                                   \
      if (cw.CP == cp && cw.NT == nt) {                                         \
        conv3x3_wmma_kernel<cp, nt><<<g, 128, 0, stream>>>(                     \
            in, cw.wf, bias, outF, outH, B, Hin, Win, cw.Cout, coutStride,      \
            Hout, Wout, stride, padBeg, relu, outMode);                         \
        return;                                                                 \
      }
    CONV_CASE(16, 1)  CONV_CASE(16, 2)  CONV_CASE(32, 4)  CONV_CASE(64, 1)
    CONV_CASE(64, 6)  CONV_CASE(96, 4)  CONV_CASE(128, 4) CONV_CASE(192, 4)
    #undef CONV_CASE
  };

  // ---- buffers: conv-facing activations are halo NHWC f16 (stride = CP of consumer)
  _Float16* h_img1 = (_Float16*)alloc((size_t)(B * Np * 16) * 2);
  _Float16* h_img2 = (_Float16*)alloc((size_t)(B * Np * 16) * 2);
  _Float16* h_f1a  = (_Float16*)alloc((size_t)(B * Np * 32) * 2);
  _Float16* h_f1b  = (_Float16*)alloc((size_t)(B * Np * 32) * 2);
  _Float16* h_f2a  = (_Float16*)alloc((size_t)(B * Nhp * 64) * 2);
  _Float16* h_f2b  = (_Float16*)alloc((size_t)(B * Nhp * 64) * 2);
  _Float16* h_f3a  = (_Float16*)alloc((size_t)(B * Nqp * 96) * 2);
  _Float16* h_f3b  = (_Float16*)alloc((size_t)(B * Nqp * 96) * 2);
  float* fsrc12 = (float*)alloc((size_t)(B * Nq * 2) * 4);
  float* fsrc21 = (float*)alloc((size_t)(B * Nq * 2) * 4);
  float* fl12z = (float*)alloc((size_t)(B * N * 2) * 4);
  float* fl21z = (float*)alloc((size_t)(B * N * 2) * 4);
  float* p12_0 = (float*)alloc((size_t)(B * N * 2) * 4);
  float* p21_0 = (float*)alloc((size_t)(B * N * 2) * 4);
  float* p12_1 = (float*)alloc((size_t)(B * Nh * 2) * 4);
  float* p21_1 = (float*)alloc((size_t)(B * Nh * 2) * 4);
  float* p12_2 = (float*)alloc((size_t)(B * Nq * 2) * 4);
  float* p21_2 = (float*)alloc((size_t)(B * Nq * 2) * 4);
  _Float16* h_mu_in = (_Float16*)alloc((size_t)(B * Np * 16) * 2);
  _Float16* h_mu_h  = (_Float16*)alloc((size_t)(B * Np * 16) * 2);
  float* m12  = (float*)alloc((size_t)(B * N) * 4);
  float* m21  = (float*)alloc((size_t)(B * N) * 4);
  float* m12h = (float*)alloc((size_t)(B * Nh) * 4);
  float* m21h = (float*)alloc((size_t)(B * Nh) * 4);
  float* m12q = (float*)alloc((size_t)(B * Nq) * 4);
  float* m21q = (float*)alloc((size_t)(B * Nq) * 4);
  float* accb = (float*)alloc((size_t)(B * N * 33) * 4);   // reused, max C=32 level
  _Float16* h_l1 = (_Float16*)alloc((size_t)(B * Np * 96) * 2);   // 70 used, pad->96
  _Float16* h_l2 = (_Float16*)alloc((size_t)(B * Nhp * 128) * 2);
  _Float16* h_l3 = (_Float16*)alloc((size_t)(B * Nqp * 192) * 2);
  float* g3 = (float*)alloc((size_t)(B * Nq * 64) * 4);
  float* g2 = (float*)alloc((size_t)(B * Nh * 64) * 4);
  float* g1 = (float*)alloc((size_t)(B * N * 64) * 4);
  _Float16* h_g1 = (_Float16*)alloc((size_t)(B * Np * 64) * 2);

  // ---- halo zeroing for conv-written buffers (producers write interiors only)
  zero_halo_kernel<<<blocks(B * (2LL*(W+2)  + 2LL*H)  * 32), 256, 0, stream>>>(h_f1a, B, H, W, 32);
  zero_halo_kernel<<<blocks(B * (2LL*(W+2)  + 2LL*H)  * 32), 256, 0, stream>>>(h_f1b, B, H, W, 32);
  zero_halo_kernel<<<blocks(B * (2LL*(W2+2) + 2LL*H2) * 64), 256, 0, stream>>>(h_f2a, B, H2, W2, 64);
  zero_halo_kernel<<<blocks(B * (2LL*(W2+2) + 2LL*H2) * 64), 256, 0, stream>>>(h_f2b, B, H2, W2, 64);
  zero_halo_kernel<<<blocks(B * (2LL*(W+2)  + 2LL*H)  * 16), 256, 0, stream>>>(h_mu_h, B, H, W, 16);
  zero_halo_kernel<<<blocks(B * (2LL*(W2+2) + 2LL*H2) * 128), 256, 0, stream>>>(h_l2, B, H2, W2, 128);
  zero_halo_kernel<<<blocks(B * (2LL*(W4+2) + 2LL*H4) * 192), 256, 0, stream>>>(h_l3, B, H4, W4, 192);
  zero_f16_kernel<<<blocks(B * Np * 96), 256, 0, stream>>>(h_l1, B * Np * 96); // ch 70..95 + halo

  // ---- images to halo NHWC f16 (border zeroed by kernel)
  img_to_nhwc16_kernel<<<blocks(B * Np), 256, 0, stream>>>(img1, h_img1, B, H, W);
  img_to_nhwc16_kernel<<<blocks(B * Np), 256, 0, stream>>>(img2, h_img2, B, H, W);

  // ---- feature pyramids (WMMA convs, ReLU, halo NHWC f16 outputs)
  conv(h_img1, cw_fe1[0], fe1b[0], nullptr, h_f1a, 32, H,  W,  H,  W,  1, 1, 1);
  conv(h_f1a,  cw_fe1[1], fe1b[1], nullptr, h_f2a, 64, H,  W,  H2, W2, 2, 1, 1);
  conv(h_f2a,  cw_fe1[2], fe1b[2], nullptr, h_f3a, 96, H2, W2, H4, W4, 2, 1, 1);
  conv(h_img2, cw_fe2[0], fe2b[0], nullptr, h_f1b, 32, H,  W,  H,  W,  1, 1, 1);
  conv(h_f1b,  cw_fe2[1], fe2b[1], nullptr, h_f2b, 64, H,  W,  H2, W2, 2, 1, 1);
  conv(h_f2b,  cw_fe2[2], fe2b[2], nullptr, h_f3b, 96, H2, W2, H4, W4, 2, 1, 1);
  // h_f3 halo must be zero for the splat reads only (never conv input, but keep clean)
  zero_halo_kernel<<<blocks(B * (2LL*(W4+2) + 2LL*H4) * 96), 256, 0, stream>>>(h_f3a, B, H4, W4, 96);
  zero_halo_kernel<<<blocks(B * (2LL*(W4+2) + 2LL*H4) * 96), 256, 0, stream>>>(h_f3b, B, H4, W4, 96);

  // ---- flow -> NHWC, fused-scale bilinear pyramids
  flow_to_nhwc_kernel<<<blocks(B * Nq), 256, 0, stream>>>(flow12, fsrc12, B, H4, W4);
  flow_to_nhwc_kernel<<<blocks(B * Nq), 256, 0, stream>>>(flow21, fsrc21, B, H4, W4);
  resize_nhwc_kernel<<<blocks(B * N * 2),  256, 0, stream>>>(fsrc12, fl12z, B, 2, H4, W4, H,  W,  20.0f, 0);
  resize_nhwc_kernel<<<blocks(B * N * 2),  256, 0, stream>>>(fsrc21, fl21z, B, 2, H4, W4, H,  W,  20.0f, 0);
  resize_nhwc_kernel<<<blocks(B * N * 2),  256, 0, stream>>>(fsrc12, p12_0, B, 2, H4, W4, H,  W,  10.0f, 0);
  resize_nhwc_kernel<<<blocks(B * N * 2),  256, 0, stream>>>(fsrc21, p21_0, B, 2, H4, W4, H,  W,  10.0f, 0);
  resize_nhwc_kernel<<<blocks(B * Nh * 2), 256, 0, stream>>>(fsrc12, p12_1, B, 2, H4, W4, H2, W2, 5.0f, 0);
  resize_nhwc_kernel<<<blocks(B * Nh * 2), 256, 0, stream>>>(fsrc21, p21_1, B, 2, H4, W4, H2, W2, 5.0f, 0);
  resize_nhwc_kernel<<<blocks(B * Nq * 2), 256, 0, stream>>>(fsrc12, p12_2, B, 2, H4, W4, H4, W4, 2.5f, 0);
  resize_nhwc_kernel<<<blocks(B * Nq * 2), 256, 0, stream>>>(fsrc21, p21_2, B, 2, H4, W4, H4, W4, 2.5f, 0);

  // ---- metric UNet (backwarp fused into input builder)
  build_mu_input_kernel<<<blocks(B * Np), 256, 0, stream>>>(img1, img2, fl12z, h_mu_in, B, H, W);
  conv(h_mu_in, cw_mu1, mu_b1, nullptr, h_mu_h, 16, H, W, H, W, 1, 1, 1);
  conv(h_mu_h,  cw_mu2, mu_b2, m12, nullptr,     1, H, W, H, W, 1, 0, 0);
  build_mu_input_kernel<<<blocks(B * Np), 256, 0, stream>>>(img2, img1, fl21z, h_mu_in, B, H, W);
  conv(h_mu_in, cw_mu1, mu_b1, nullptr, h_mu_h, 16, H, W, H, W, 1, 1, 1);
  conv(h_mu_h,  cw_mu2, mu_b2, m21, nullptr,     1, H, W, H, W, 1, 0, 0);

  resize_nhwc_kernel<<<blocks(B * Nh), 256, 0, stream>>>(m12, m12h, B, 1, H, W, H2, W2, 1.0f, 0);
  resize_nhwc_kernel<<<blocks(B * Nq), 256, 0, stream>>>(m12, m12q, B, 1, H, W, H4, W4, 1.0f, 0);
  resize_nhwc_kernel<<<blocks(B * Nh), 256, 0, stream>>>(m21, m21h, B, 1, H, W, H2, W2, 1.0f, 0);
  resize_nhwc_kernel<<<blocks(B * Nq), 256, 0, stream>>>(m21, m21q, B, 1, H, W, H4, W4, 1.0f, 0);

  // ---- softmax splatting into concatenated halo NHWC GridNet inputs
  auto splat = [&](const _Float16* inp, int Cs, const float* flow, const float* metric,
                   const float* beta, int C, int Hl, int Wl,
                   _Float16* dst, int Ctot, int chOff) {
    long long pn = (long long)B * Hl * Wl;
    long long an = pn * (C + 1);
    zero_f32_kernel<<<blocks(an), 256, 0, stream>>>(accb, an);
    softsplat_scatter_kernel<<<blocks(pn), 256, 0, stream>>>(inp, flow, metric, beta, accb, B, C, Cs, Hl, Wl);
    softsplat_norm_kernel<<<blocks(pn * C), 256, 0, stream>>>(accb, dst, B, C, Hl, Wl, Ctot, chOff);
  };
  splat(h_img1, 16, p12_0, m12,  beta1,  3, H,  W,  h_l1, 96,  0);
  splat(h_f1a,  32, p12_0, m12,  beta1, 32, H,  W,  h_l1, 96,  3);
  splat(h_img2, 16, p21_0, m21,  beta2,  3, H,  W,  h_l1, 96, 35);
  splat(h_f1b,  32, p21_0, m21,  beta2, 32, H,  W,  h_l1, 96, 38);
  splat(h_f2a,  64, p12_1, m12h, beta1, 64, H2, W2, h_l2, 128,  0);
  splat(h_f2b,  64, p21_1, m21h, beta2, 64, H2, W2, h_l2, 128, 64);
  splat(h_f3a,  96, p12_2, m12q, beta1, 96, H4, W4, h_l3, 192,  0);
  splat(h_f3b,  96, p21_2, m21q, beta2, 96, H4, W4, h_l3, 192, 96);

  // ---- GridNet: coarse-to-fine fusion (f32 linear NHWC intermediates)
  conv(h_l3, cw_gn3, gn_b3, g3, nullptr, 64, H4, W4, H4, W4, 1, 1, 0);
  conv(h_l2, cw_gn2, gn_b2, g2, nullptr, 64, H2, W2, H2, W2, 1, 1, 0);
  resize_nhwc_kernel<<<blocks(B * Nh * 64), 256, 0, stream>>>(g3, g2, B, 64, H4, W4, H2, W2, 1.0f, 1);
  conv(h_l1, cw_gn1, gn_b1, g1, nullptr, 64, H, W, H, W, 1, 1, 0);
  resize_nhwc_kernel<<<blocks(B * N * 64), 256, 0, stream>>>(g2, g1, B, 64, H2, W2, H, W, 1.0f, 1);
  f32_to_f16_halo_kernel<<<blocks(B * Np * 64), 256, 0, stream>>>(g1, h_g1, B, H, W, 64);
  conv(h_g1, cw_gnf, gn_bf, (float*)d_out, nullptr, 3, H, W, H, W, 1, 0, 2);  // NCHW f32
}
